// Memory_3358664425627
// MI455X (gfx1250) — matrix operand order repaired and checked by reading, not compiled, not run on hardware
//
#include <hip/hip_runtime.h>
#include <hip/hip_bf16.h>
#include <stdint.h>

// ---------------------------------------------------------------------------
// Memory-network kernel for MI455X (gfx1250, wave32, WMMA).
// Pipeline: gemm1+relu -> gemm2 -> batchnorm-stats -> wmma scores + top-k +
// softmax -> async-LDS gather / weighted-sum / batch-reduce.
// Matrix math: v_wmma_f32_16x16x32_bf16 (f32 accumulate).
// Data movement: global_load_async_to_lds_b128 (ASYNCcnt) for the gather,
// global_prefetch_b8 for GEMM streams.
// ---------------------------------------------------------------------------

typedef __attribute__((ext_vector_type(16))) __bf16 v16bf;
typedef __attribute__((ext_vector_type(8)))  float  v8f;

#define NEG_INF (-1e30f)

// ---------------------------------------------------------------------------
// Generic GEMM: out[m,n] = act( sum_k A[m,k] * W[n,k] + bias[n] )
// One wave (32 threads) computes one 16x16 tile of C.
// A layout (16-bit, 16x32):  lane&15 = M row; lane>>4 selects K sub-chunks
//   v[0..7]  = K = (lane>>4)*8 + 0..7
//   v[8..15] = K = 16 + (lane>>4)*8 + 0..7
// B layout (32x16):          lane&15 = N col; lane>>4 selects K sub-chunks (same as A)
//   v[0..7] = K = (lane>>4)*8 + 0..7, v[8..15] = K = 16 + (lane>>4)*8 + 0..7
// C layout: VGPR i -> M = i + (lane>>4)*8, N = lane&15
// ---------------------------------------------------------------------------
template <int K, int N, bool RELU>
__global__ void k_gemm(const float* __restrict__ A, const float* __restrict__ W,
                       const float* __restrict__ bias, float* __restrict__ out) {
    const int lane = threadIdx.x;        // 32-thread block
    const int m0 = blockIdx.x * 16;
    const int n0 = blockIdx.y * 16;
    const int r   = lane & 15;
    const int hi  = lane >> 4;
    const int kgA = hi * 8;
    const int kgB = hi * 16;

    v8f acc = {};
    const float* ap = A + (size_t)(m0 + r) * K;
    const float* wp = W + (size_t)(n0 + r) * K;

    for (int k0 = 0; k0 < K; k0 += 32) {
        if (k0 + 32 < K) {                       // prefetch next K step
            __builtin_prefetch(ap + k0 + 32 + kgA, 0, 0);
            __builtin_prefetch(wp + k0 + 32 + kgB, 0, 0);
        }
        v16bf a, b;
#pragma unroll
        for (int i = 0; i < 8; ++i) {
            a[i]     = (__bf16)ap[k0 + kgA + i];
            a[i + 8] = (__bf16)ap[k0 + 16 + kgA + i];
        }
#pragma unroll
        for (int i = 0; i < 16; ++i) b[i] = (__bf16)wp[k0 + (i >> 3) * 16 + kgA + (i & 7)];
        acc = __builtin_amdgcn_wmma_f32_16x16x32_bf16(
            false, a, false, b, (short)0, acc, false, false);
    }

    const int col = n0 + r;
    const float bv = bias[col];
    const int mb = m0 + hi * 8;
#pragma unroll
    for (int i = 0; i < 8; ++i) {
        float v = acc[i] + bv;
        if (RELU) v = v > 0.f ? v : 0.f;
        out[(size_t)(mb + i) * N + col] = v;
    }
}

// ---------------------------------------------------------------------------
// BatchNorm statistics over q viewed as (8192, 256); fold gamma/beta:
//   scale[c] = gamma[c] * rsqrt(var+eps);  shift[c] = beta[c] - mean*scale[c]
// One block (256 threads) per channel.
// ---------------------------------------------------------------------------
__global__ void k_bn_stats(const float* __restrict__ q,
                           const float* __restrict__ gamma,
                           const float* __restrict__ beta,
                           float* __restrict__ scale,
                           float* __restrict__ shift) {
    __shared__ float rs[256], rq[256];
    const int c = blockIdx.x;
    const int tid = threadIdx.x;
    float a = 0.f, sq = 0.f;
    for (int f = tid; f < 8192; f += 256) {
        float v = q[(size_t)f * 256 + c];
        a += v;
        sq += v * v;
    }
    rs[tid] = a;
    rq[tid] = sq;
    __syncthreads();
    for (int o = 128; o > 0; o >>= 1) {
        if (tid < o) { rs[tid] += rs[tid + o]; rq[tid] += rq[tid + o]; }
        __syncthreads();
    }
    if (tid == 0) {
        float mean = rs[0] * (1.f / 8192.f);
        float var  = rq[0] * (1.f / 8192.f) - mean * mean;
        float sc   = gamma[c] * rsqrtf(var + 1e-5f);
        scale[c] = sc;
        shift[c] = beta[c] - mean * sc;
    }
}

// ---------------------------------------------------------------------------
// Scores + two-level top-k + softmax.
// Block = 256 threads (8 waves), grid = (2048/16 token tiles, 4 heads).
// Per half: waves compute 16x512 score tile into 32KB LDS via WMMA
// (normalize-on-load of q, bf16 convert), then each wave does iterative
// top-32-of-512 for its 2 tokens. LDS accessed by direct indexing only
// (ds_load/ds_store; the masking store aliases the array, forcing reloads —
// no volatile, so no flat SYS-scope fallback). Then the 32x32 pair combine
// and softmax run purely on shuffles.
// ---------------------------------------------------------------------------
__global__ void k_score_topk(const float* __restrict__ q,
                             const float* __restrict__ scale,
                             const float* __restrict__ shift,
                             const float* __restrict__ keys,
                             int* __restrict__ topk_idx,
                             float* __restrict__ topk_w) {
    __shared__ float s[16][512];          // 32 KB
    const int lane = threadIdx.x & 31;
    const int wave = threadIdx.x >> 5;    // 0..7
    const int m0   = blockIdx.x * 16;
    const int head = blockIdx.y;
    const int r    = lane & 15;
    const int hi   = lane >> 4;
    const int kgA  = hi * 8;
    const int kgB  = hi * 16;

    float ts1[2], ts2[2];
    int   ti1[2], ti2[2];

    for (int half = 0; half < 2; ++half) {
        // ---- phase A: 16x512 score tile for this half ----
        const float* qrow = q + (size_t)(m0 + r) * 1024 + head * 256 + half * 128;
        const int cb = half * 128;
        for (int t = 0; t < 4; ++t) {
            const int key0 = (wave * 4 + t) * 16;
            const float* krow =
                keys + ((size_t)((head * 2 + half) * 512 + key0 + r)) * 128;
            v8f acc = {};
            for (int k0 = 0; k0 < 128; k0 += 32) {
                v16bf a, b;
#pragma unroll
                for (int i = 0; i < 8; ++i) {
                    const int c0 = k0 + kgA + i;
                    const int c1 = k0 + 16 + kgA + i;
                    a[i]     = (__bf16)(qrow[c0] * scale[cb + c0] + shift[cb + c0]);
                    a[i + 8] = (__bf16)(qrow[c1] * scale[cb + c1] + shift[cb + c1]);
                }
#pragma unroll
                for (int i = 0; i < 16; ++i) b[i] = (__bf16)krow[k0 + (i >> 3) * 16 + kgA + (i & 7)];
                acc = __builtin_amdgcn_wmma_f32_16x16x32_bf16(
                    false, a, false, b, (short)0, acc, false, false);
            }
#pragma unroll
            for (int i = 0; i < 8; ++i) s[i + hi * 8][key0 + r] = acc[i];
        }
        __syncthreads();

        // ---- phase B: top-32 of 512 for this wave's 2 tokens ----
        for (int t = 0; t < 2; ++t) {
            const int row = wave * 2 + t;
            float my_s = NEG_INF;
            int   my_i = 0;
            for (int k = 0; k < 32; ++k) {
                float best = NEG_INF;
                int   bidx = 0;
                for (int e = lane; e < 512; e += 32) {
                    float v = s[row][e];               // ds_load_b32
                    if (v > best) { best = v; bidx = e; }
                }
                for (int off = 16; off > 0; off >>= 1) {
                    float ov = __shfl_xor(best, off, 32);
                    int   oi = __shfl_xor(bidx, off, 32);
                    if (ov > best) { best = ov; bidx = oi; }
                }
                if (lane == k) { my_s = best; my_i = bidx; }
                if (lane == 0) s[row][bidx] = NEG_INF; // mask winner (ds_store)
            }
            if (half == 0) { ts1[t] = my_s; ti1[t] = my_i; }
            else           { ts2[t] = my_s; ti2[t] = my_i; }
        }
        __syncthreads();   // all waves done before half-1 overwrites LDS
    }

    // ---- combine 32x32 pair sums, top-32-of-1024, softmax ----
    for (int t = 0; t < 2; ++t) {
        const int n = m0 + wave * 2 + t;
        unsigned used = 0;            // lane i: mask of consumed j's for row i
        float sc = NEG_INF;
        int   id = 0;
        for (int k = 0; k < 32; ++k) {
            float best = NEG_INF;
            int   bi = lane, bj = 0;
            for (int j = 0; j < 32; ++j) {
                float t2 = __shfl(ts2[t], j, 32);     // uniform exec
                float v  = ts1[t] + t2;
                if (!((used >> j) & 1u) && v > best) { best = v; bj = j; }
            }
            for (int off = 16; off > 0; off >>= 1) {
                float ov  = __shfl_xor(best, off, 32);
                int   obi = __shfl_xor(bi, off, 32);
                int   obj = __shfl_xor(bj, off, 32);
                if (ov > best) { best = ov; bi = obi; bj = obj; }
            }
            if (lane == bi) used |= (1u << bj);
            const int wi1 = __shfl(ti1[t], bi, 32);
            const int wi2 = __shfl(ti2[t], bj, 32);
            if (lane == k) { sc = best; id = wi1 * 512 + wi2; }
        }
        // softmax across the 32 lanes (lane k holds rank-k score)
        float m = sc;
        for (int off = 16; off > 0; off >>= 1) m = fmaxf(m, __shfl_xor(m, off, 32));
        float e = __expf(sc - m);
        float ssum = e;
        for (int off = 16; off > 0; off >>= 1) ssum += __shfl_xor(ssum, off, 32);

        const size_t base = ((size_t)n * 4 + head) * 32;
        topk_idx[base + lane] = id;
        topk_w[base + lane]   = e / ssum;
    }
}

// ---------------------------------------------------------------------------
// Gather + weighted sum + batch reduce: out[b,:] += sum_k w * values[idx,:]
// Block per (token, head), 128 threads. The 32 gathered rows (2 KB each) are
// staged VMEM->LDS with global_load_async_to_lds_b128 in two 32KB batches
// (GVS mode: SGPR base = values, VGPR byte offset), ASYNCcnt-tracked, then
// reduced out of LDS with ds_load_b128.
// ---------------------------------------------------------------------------
__global__ void k_gather(const float* __restrict__ values,
                         const int* __restrict__ topk_idx,
                         const float* __restrict__ topk_w,
                         float* __restrict__ out) {
    __shared__ float vrow[16][512];     // 32 KB staging
    __shared__ int   sidx[32];
    __shared__ float sw[32];
    const int n = blockIdx.x, head = blockIdx.y;
    const int b = n >> 6;                       // n = b*64 + l
    const size_t base = ((size_t)n * 4 + head) * 32;
    const int tid = threadIdx.x;                // 128 threads
    const int d = tid * 4;                      // float index, 16B aligned

    if (tid < 32) {
        sidx[tid] = topk_idx[base + tid];
        sw[tid]   = topk_w[base + tid];
    }
    __syncthreads();

    float ax = 0.f, ay = 0.f, az = 0.f, aw = 0.f;
    for (int batch = 0; batch < 2; ++batch) {
#pragma unroll
        for (int k = 0; k < 16; ++k) {
            const unsigned row  = (unsigned)sidx[batch * 16 + k];
            const unsigned voff = row * 2048u + (unsigned)d * 4u; // byte offset
            const unsigned ldst = (unsigned)(uintptr_t)&vrow[k][d];
            asm volatile("global_load_async_to_lds_b128 %0, %1, %2"
                         :: "v"(ldst), "v"(voff), "s"(values)
                         : "memory");
        }
        asm volatile("s_wait_asynccnt 0" ::: "memory");
        __syncthreads();
#pragma unroll
        for (int k = 0; k < 16; ++k) {
            const float w = sw[batch * 16 + k];
            const float4 v = *reinterpret_cast<const float4*>(&vrow[k][d]);
            ax += w * v.x; ay += w * v.y; az += w * v.z; aw += w * v.w;
        }
        __syncthreads();
    }
    float* o = out + (size_t)b * 512 + d;
    atomicAdd(o + 0, ax);
    atomicAdd(o + 1, ay);
    atomicAdd(o + 2, az);
    atomicAdd(o + 3, aw);
}

// ---------------------------------------------------------------------------
// Launch. Workspace layout (needs ~12.6 MB):
//   [0, 2MB)            h     (2048 x 256 f32)
//   [2MB, 10MB)         q     (2048 x 1024 f32)
//   [10MB, +1KB)        scale (256 f32)
//   [+1KB, +2KB)        shift (256 f32)
//   [10MB+4KB, +1MB)    topk_idx (2048*4*32 i32)
//   [.. +1MB)           topk_w   (2048*4*32 f32)
// ---------------------------------------------------------------------------
extern "C" void kernel_launch(void* const* d_in, const int* in_sizes, int n_in,
                              void* d_out, int out_size, void* d_ws,
                              size_t ws_size, hipStream_t stream) {
    const float* x      = (const float*)d_in[0];
    const float* w1     = (const float*)d_in[1];
    const float* b1     = (const float*)d_in[2];
    const float* w2     = (const float*)d_in[3];
    const float* b2     = (const float*)d_in[4];
    const float* gamma  = (const float*)d_in[5];
    const float* beta   = (const float*)d_in[6];
    const float* keys   = (const float*)d_in[7];
    const float* values = (const float*)d_in[8];
    float* out = (float*)d_out;

    char* ws = (char*)d_ws;
    float* h     = (float*)(ws);
    float* qbuf  = (float*)(ws + (2u << 20));
    float* scale = (float*)(ws + (10u << 20));
    float* shift = scale + 256;
    int*   tidx  = (int*)(ws + (10u << 20) + 4096);
    float* tw    = (float*)(ws + (10u << 20) + 4096 + (1u << 20));

    hipMemsetAsync(d_out, 0, (size_t)32 * 512 * sizeof(float), stream);

    // h = relu(x @ w1^T + b1)   : M=2048, K=512, N=256
    k_gemm<512, 256, true><<<dim3(128, 16), 32, 0, stream>>>(x, w1, b1, h);
    // q = h @ w2^T + b2         : M=2048, K=256, N=1024
    k_gemm<256, 1024, false><<<dim3(128, 64), 32, 0, stream>>>(h, w2, b2, qbuf);
    // BN stats folded into scale/shift
    k_bn_stats<<<256, 256, 0, stream>>>(qbuf, gamma, beta, scale, shift);
    // normalized WMMA scores + two-level top-k + softmax
    k_score_topk<<<dim3(128, 4), 256, 0, stream>>>(qbuf, scale, shift, keys,
                                                   tidx, tw);
    // async-LDS gather + weighted sum + batch reduce
    k_gather<<<dim3(2048, 4), 128, 0, stream>>>(values, tidx, tw, out);
}
